// MonotonicSeq2Seq_9440338116609
// MI455X (gfx1250) — matrix operand order repaired and checked by reading, not compiled, not run on hardware
//
#include <hip/hip_runtime.h>
#include <hip/hip_bf16.h>

#define B_   16
#define TQ_  512
#define TK_  512
#define V_   10000
#define D_   512
#define NEG_ (-1e9f)

typedef __attribute__((ext_vector_type(16))) __bf16 bfvec16;
typedef __attribute__((ext_vector_type(8)))  float  fvec8;

union FragU { bfvec16 v; uint4 u[2]; };

__device__ __forceinline__ __bf16 f2bf(float x) { return (__bf16)x; }

// ---------------------------------------------------------------------------
// A-fragment (16x32 bf16): lane holds row M=lane&15; K chunks {kb..kb+7} and
// {kb+16..kb+23}, kb = 8*(lane>=16). Two 16B loads.
// ---------------------------------------------------------------------------
__device__ __forceinline__ bfvec16 load_a_frag(const __bf16* base, int ld,
                                               int row, int k0, int lane) {
  int kb = k0 + ((lane >> 4) << 3);
  const __bf16* p = base + (size_t)row * ld + kb;
  FragU f;
  f.u[0] = *reinterpret_cast<const uint4*>(p);
  f.u[1] = *reinterpret_cast<const uint4*>(p + 16);
  return f.v;
}

// ---------------------------------------------------------------------------
// B-fragment (32x16 bf16) from transposed (N,K) storage: lane holds column
// N=lane&15; K chunks {kb..kb+7} and {kb+16..kb+23}, kb = k0 + 8*(lane>=16).
// ---------------------------------------------------------------------------
__device__ __forceinline__ bfvec16 load_b_frag(const __bf16* baseT, int ld,
                                               int col, int k0, int lane) {
  int kb = k0 + ((lane >> 4) << 3);
  const __bf16* p = baseT + (size_t)col * ld + kb;
  FragU f;
  f.u[0] = *reinterpret_cast<const uint4*>(p);
  f.u[1] = *reinterpret_cast<const uint4*>(p + 16);
  return f.v;
}

// ---------------------------------------------------------------------------
// Tile body templated on JN (# of valid 16-col sub-tiles, 1..4) so the hot
// path (JN==4) has a completely branch-free K-loop: 2 A-frag loads + JN
// B-frag loads + 2*JN WMMAs per 32-deep K step.
// ---------------------------------------------------------------------------
template <int JN>
__device__ __forceinline__ void gemm_tile(const __bf16* __restrict__ Ab, int lda,
                                          const __bf16* __restrict__ Bb, int ldb,
                                          const float* __restrict__ bias,
                                          void* __restrict__ C, long long cBase,
                                          int ldc, int K, int mode, float scale,
                                          int tm, int n0, int lane) {
  const int rowA0 = (tm << 5) + (lane & 15);
  const int rowA1 = rowA0 + 16;

  fvec8 acc[2][JN] = {};
  for (int k0 = 0; k0 < K; k0 += 32) {
    const bfvec16 a0 = load_a_frag(Ab, lda, rowA0, k0, lane);
    const bfvec16 a1 = load_a_frag(Ab, lda, rowA1, k0, lane);
#pragma unroll
    for (int j = 0; j < JN; ++j) {
      const bfvec16 bfr =
          load_b_frag(Bb, ldb, n0 + j * 16 + (lane & 15), k0, lane);
      acc[0][j] = __builtin_amdgcn_wmma_f32_16x16x32_bf16(
          false, a0, false, bfr, (short)0, acc[0][j], false, false);
      acc[1][j] = __builtin_amdgcn_wmma_f32_16x16x32_bf16(
          false, a1, false, bfr, (short)0, acc[1][j], false, false);
    }
  }

  // C/D layout: lane N=lane&15; VGPR r -> M = r + 8*(lane>=16)
  const int nC    = lane & 15;
  const int mHalf = (lane >> 4) << 3;
#pragma unroll
  for (int h = 0; h < 2; ++h) {
    const int mTile = (tm << 5) + h * 16 + mHalf;
#pragma unroll
    for (int j = 0; j < JN; ++j) {
      const int ncol = n0 + j * 16 + nC;
      const float bv = bias ? bias[ncol] : 0.0f;
#pragma unroll
      for (int r = 0; r < 8; ++r) {
        const float val = acc[h][j][r] + bv;
        const int mrow = mTile + r;
        if (mode == 0) {
          ((__bf16*)C)[cBase + (long long)mrow * ldc + ncol] = f2bf(val);
        } else if (mode == 1) {
          ((__bf16*)C)[cBase + (long long)ncol * ldc + mrow] = f2bf(val);
        } else if (mode == 2) {
          ((float*)C)[cBase + (long long)ncol * ldc + mrow] = val * scale;
        } else {
          ((float*)C)[cBase + (long long)mrow * ldc + ncol] = val;
        }
      }
    }
  }
}

// ---------------------------------------------------------------------------
// Generic batched GEMM: C = A(M,K) @ Bt(N,K)^T [+ bias], bf16 in, f32 accum.
// Each wave owns a 32x64 strip. All tile coordinates are forced into SGPRs
// via readfirstlane so control flow is uniform and WMMA runs with full EXEC.
// mode 0: store bf16 C(M,N)            mode 1: store bf16 C^T(N,M)
// mode 2: store f32  C^T(N,M)*scale    mode 3: store f32  C(M,N)
// ---------------------------------------------------------------------------
__global__ __launch_bounds__(256)
void wmma_gemm_bf16(const __bf16* __restrict__ A, long long sAb, int lda,
                    const __bf16* __restrict__ Bt, long long sBb, int ldb,
                    const float* __restrict__ bias,
                    void* __restrict__ C, long long sCb, int ldc,
                    int batch, int M, int N, int K, int mode, float scale) {
  const int lane = threadIdx.x & 31;
  // wave id as a scalar (SGPR) -> uniform control flow, EXEC stays all-1s
  const int wvid = __builtin_amdgcn_readfirstlane((int)(threadIdx.x >> 5));

  const int tilesM2 = M >> 5;            // 32-row strips
  const int tilesN4 = (N + 63) >> 6;     // 64-col groups
  const long long tilesPerB = (long long)tilesM2 * tilesN4;
  const long long tile = (long long)blockIdx.x * 8 + wvid;
  if (tile >= tilesPerB * (long long)batch) return;

  const int b   = (int)(tile / tilesPerB);
  const int t   = (int)(tile % tilesPerB);
  const int tm  = t / tilesN4;
  const int tn4 = t % tilesN4;
  const int n0  = tn4 << 6;
  const int nsub = ((N - n0) >= 64) ? 4 : ((N - n0) >> 4);  // scalar, 1..4

  const __bf16* Ab = A  + (long long)b * sAb;
  const __bf16* Bb = Bt + (long long)b * sBb;
  const long long cBase = (long long)b * sCb;

  switch (nsub) {  // uniform dispatch; each body is branch-free inside
    case 4: gemm_tile<4>(Ab, lda, Bb, ldb, bias, C, cBase, ldc, K, mode, scale, tm, n0, lane); break;
    case 3: gemm_tile<3>(Ab, lda, Bb, ldb, bias, C, cBase, ldc, K, mode, scale, tm, n0, lane); break;
    case 2: gemm_tile<2>(Ab, lda, Bb, ldb, bias, C, cBase, ldc, K, mode, scale, tm, n0, lane); break;
    default: gemm_tile<1>(Ab, lda, Bb, ldb, bias, C, cBase, ldc, K, mode, scale, tm, n0, lane); break;
  }
}

// ---------------------------------------------------------------------------
// Embedding gathers: src_e (B,TK,D), tgt_e (B,TQ,D) as bf16; PAD(0) row -> 0;
// tgt_e row 0 = init_emb, row q = emb[tgt[q-1]].
// ---------------------------------------------------------------------------
__global__ __launch_bounds__(256)
void gather_embed(const int* __restrict__ src, const int* __restrict__ tgt,
                  const float* __restrict__ emb, const float* __restrict__ init_emb,
                  __bf16* __restrict__ src_e, __bf16* __restrict__ tgt_e) {
  const long long i = (long long)blockIdx.x * blockDim.x + threadIdx.x;
  const long long total = (long long)B_ * TK_ * D_;   // == B_*TQ_*D_
  if (i >= total) return;
  const int d = (int)(i % D_);
  const long long bt = i / D_;
  const int t = (int)(bt % TK_);
  const int b = (int)(bt / TK_);

  { // src side
    const int tok = src[b * TK_ + t];
    const float x = (tok == 0) ? 0.0f : emb[(long long)tok * D_ + d];
    src_e[i] = f2bf(x);
  }
  { // tgt side (shifted, init_emb at q==0)
    float x;
    if (t == 0) x = init_emb[d];
    else {
      const int tok = tgt[b * TQ_ + (t - 1)];
      x = (tok == 0) ? 0.0f : emb[(long long)tok * D_ + d];
    }
    tgt_e[i] = f2bf(x);
  }
}

// ---------------------------------------------------------------------------
// Weight transpose + fp32 -> bf16: WT(N,K)[n*K+k] = W(K,N)[k*N+n]
// ---------------------------------------------------------------------------
__global__ __launch_bounds__(256)
void transpose_to_bf16(const float* __restrict__ W, __bf16* __restrict__ WT,
                       int Kd, int Nd) {
  const long long i = (long long)blockIdx.x * blockDim.x + threadIdx.x;
  if (i >= (long long)Kd * Nd) return;
  const int k = (int)(i % Kd);
  const int n = (int)(i / Kd);
  WT[i] = f2bf(W[(long long)k * Nd + n]);
}

// ---------------------------------------------------------------------------
// Monotonic lattice scan (log space), one block per batch, thread = query q.
// Reads eT (B,TK,TQ) fp32 columns coalesced; writes attn (B,TQ,TK) bf16.
// phi(q,j) = logaddexp(phi(q,j-1)+log_sig(e), phi(q-1,j-1)+log_sig(-e_{q-1}))
// ---------------------------------------------------------------------------
__global__ __launch_bounds__(TQ_)
void mono_scan(const float* __restrict__ eT, __bf16* __restrict__ attn) {
  const int b = blockIdx.x;
  const int q = threadIdx.x;
  __shared__ float advs[TQ_];

  float ph = (q == 0) ? 0.0f : NEG_;
  attn[((long long)b * TQ_ + q) * TK_ + 0] = f2bf(__expf(ph));
  const float* ecol = eT + (long long)b * TK_ * TQ_;

  for (int j = 1; j < TK_; ++j) {
    const float e = ecol[(long long)(j - 1) * TQ_ + q];
    const float ae = fabsf(e);
    const float l1pe = log1pf(__expf(-ae));
    const float lp   = fminf(e, 0.0f) - l1pe;   // log sigmoid(e)
    const float l1mp = fminf(-e, 0.0f) - l1pe;  // log sigmoid(-e)
    advs[q] = ph + l1mp;
    __syncthreads();
    const float stay = ph + lp;
    const float adv  = (q == 0) ? NEG_ : advs[q - 1];
    const float mx = fmaxf(stay, adv);
    const float mn = fminf(stay, adv);
    ph = mx + log1pf(__expf(mn - mx));
    attn[((long long)b * TQ_ + q) * TK_ + j] = f2bf(__expf(ph));
    __syncthreads();
  }
}

// ---------------------------------------------------------------------------
extern "C" void kernel_launch(void* const* d_in, const int* in_sizes, int n_in,
                              void* d_out, int out_size, void* d_ws, size_t ws_size,
                              hipStream_t stream) {
  const int*   src      = (const int*)  d_in[0];
  const int*   tgt      = (const int*)  d_in[1];
  const float* emb      = (const float*)d_in[2];
  const float* init_emb = (const float*)d_in[3];
  const float* wq = (const float*)d_in[4];  const float* bq = (const float*)d_in[5];
  const float* wk = (const float*)d_in[6];  const float* bk = (const float*)d_in[7];
  const float* wvp= (const float*)d_in[8];  const float* bv = (const float*)d_in[9];
  const float* wo = (const float*)d_in[10]; const float* bo = (const float*)d_in[11];
  const float* wp = (const float*)d_in[12]; const float* bp = (const float*)d_in[13];
  float* out = (float*)d_out;

  // workspace carve-up (256B aligned)
  char* cur = (char*)d_ws;
  auto alloc = [&](size_t bytes) {
    char* r = cur;
    cur += (bytes + 255) & ~(size_t)255;
    return r;
  };
  __bf16* tgt_e = (__bf16*)alloc((size_t)B_ * TQ_ * D_ * 2);
  __bf16* src_e = (__bf16*)alloc((size_t)B_ * TK_ * D_ * 2);
  __bf16* wqT   = (__bf16*)alloc((size_t)D_ * D_ * 2);
  __bf16* wkT   = (__bf16*)alloc((size_t)D_ * D_ * 2);
  __bf16* wvT   = (__bf16*)alloc((size_t)D_ * D_ * 2);
  __bf16* woT   = (__bf16*)alloc((size_t)D_ * D_ * 2);
  __bf16* wpT   = (__bf16*)alloc((size_t)V_ * D_ * 2);
  __bf16* qb    = (__bf16*)alloc((size_t)B_ * TQ_ * D_ * 2);
  __bf16* kb    = (__bf16*)alloc((size_t)B_ * TK_ * D_ * 2);
  __bf16* vT    = (__bf16*)alloc((size_t)B_ * D_ * TK_ * 2);   // (B, D, TK)
  float*  eT    = (float*) alloc((size_t)B_ * TK_ * TQ_ * 4);  // (B, TK, TQ)
  __bf16* attn  = (__bf16*)alloc((size_t)B_ * TQ_ * TK_ * 2);
  __bf16* xb    = (__bf16*)alloc((size_t)B_ * TQ_ * D_ * 2);
  __bf16* hb    = (__bf16*)alloc((size_t)B_ * TQ_ * D_ * 2);

  const int BT = B_ * TQ_;  // 8192 rows for fused-batch GEMMs
  auto gemm_blocks = [](int batch, int M, int N) {
    long long tiles = (long long)batch * (M >> 5) * ((N + 63) >> 6);
    return (int)((tiles + 7) / 8);
  };

  // 1) embedding gathers
  {
    long long total = (long long)B_ * TK_ * D_;
    gather_embed<<<(int)((total + 255) / 256), 256, 0, stream>>>(
        src, tgt, emb, init_emb, src_e, tgt_e);
  }
  // 2) weight transposes to bf16 (N,K)
  {
    int nDD = (int)(((long long)D_ * D_ + 255) / 256);
    transpose_to_bf16<<<nDD, 256, 0, stream>>>(wq,  wqT, D_, D_);
    transpose_to_bf16<<<nDD, 256, 0, stream>>>(wk,  wkT, D_, D_);
    transpose_to_bf16<<<nDD, 256, 0, stream>>>(wvp, wvT, D_, D_);
    transpose_to_bf16<<<nDD, 256, 0, stream>>>(wo,  woT, D_, D_);
    int nDV = (int)(((long long)D_ * V_ + 255) / 256);
    transpose_to_bf16<<<nDV, 256, 0, stream>>>(wp,  wpT, D_, V_);
  }
  // 3) q = tgt_e @ wq + bq   (bf16 out)
  wmma_gemm_bf16<<<gemm_blocks(1, BT, D_), 256, 0, stream>>>(
      tgt_e, 0, D_, wqT, 0, D_, bq, qb, 0, D_, 1, BT, D_, D_, 0, 1.0f);
  // 4) k = src_e @ wk + bk   (bf16 out)
  wmma_gemm_bf16<<<gemm_blocks(1, BT, D_), 256, 0, stream>>>(
      src_e, 0, D_, wkT, 0, D_, bk, kb, 0, D_, 1, BT, D_, D_, 0, 1.0f);
  // 5) v = src_e @ wv + bv, stored transposed per batch: vT (B, D, TK)
  wmma_gemm_bf16<<<gemm_blocks(B_, TK_, D_), 256, 0, stream>>>(
      src_e, (long long)TK_ * D_, D_, wvT, 0, D_, bv,
      vT, (long long)D_ * TK_, TK_, B_, TK_, D_, D_, 1, 1.0f);
  // 6) eT = (q @ k^T / sqrt(D))^T per batch: (B, TK, TQ) fp32
  wmma_gemm_bf16<<<gemm_blocks(B_, TQ_, TK_), 256, 0, stream>>>(
      qb, (long long)TQ_ * D_, D_, kb, (long long)TK_ * D_, D_, nullptr,
      eT, (long long)TK_ * TQ_, TQ_, B_, TQ_, TK_, D_, 2, 0.044194173824159216f);
  // 7) monotonic lattice scan -> attn (B,TQ,TK) bf16
  mono_scan<<<B_, TQ_, 0, stream>>>(eT, attn);
  // 8) x = attn @ v  (A=(TQ,TK), Bt=vT(D,TK)) -> bf16 (B,TQ,D)
  wmma_gemm_bf16<<<gemm_blocks(B_, TQ_, D_), 256, 0, stream>>>(
      attn, (long long)TQ_ * TK_, TK_, vT, (long long)D_ * TK_, TK_, nullptr,
      xb, (long long)TQ_ * D_, D_, B_, TQ_, D_, TK_, 0, 1.0f);
  // 9) h = x @ wo + bo  -> bf16
  wmma_gemm_bf16<<<gemm_blocks(1, BT, D_), 256, 0, stream>>>(
      xb, 0, D_, woT, 0, D_, bo, hb, 0, D_, 1, BT, D_, D_, 0, 1.0f);
  // 10) out = h @ wp + bp  -> fp32 (B,TQ,V)
  wmma_gemm_bf16<<<gemm_blocks(1, BT, V_), 256, 0, stream>>>(
      hb, 0, D_, wpT, 0, D_, bp, out, 0, V_, 1, BT, V_, D_, 3, 1.0f);

  (void)in_sizes; (void)n_in; (void)out_size; (void)ws_size;
}
